// AdvSmax_63522566308202
// MI455X (gfx1250) — compile-verified, matrix operand order checked by reading, not run on hardware
//
#include <hip/hip_runtime.h>
#include <math.h>

#define NTOK   50257
#define DK     400
#define DKP    416          // K padded to multiple of 32
#define NROW   2240         // 35*64
#define KSTEPS 13           // 416/32
#define ALPHA  0.2f
#define EPSN   1e-8f

typedef __attribute__((ext_vector_type(16))) __bf16 v16bf;
typedef __attribute__((ext_vector_type(8)))  __bf16 v8bf;
typedef __attribute__((ext_vector_type(8)))  float  v8f;
typedef __attribute__((ext_vector_type(4)))  float  v4f;

// ---------------- bf16 split helpers (truncation split: x = hi + lo + O(2^-16)) ---
__device__ __forceinline__ __bf16 f2bf_trunc(float x) {
  unsigned u = __builtin_bit_cast(unsigned, x);
  unsigned short s = (unsigned short)(u >> 16);
  return __builtin_bit_cast(__bf16, s);
}
__device__ __forceinline__ float bf2f(__bf16 b) {
  unsigned u = ((unsigned)__builtin_bit_cast(unsigned short, b)) << 16;
  return __builtin_bit_cast(float, u);
}
__device__ __forceinline__ void split2(float x, __bf16 &hi, __bf16 &lo) {
  hi = f2bf_trunc(x);
  lo = f2bf_trunc(x - bf2f(hi));
}

// ---------------- workspace init -------------------------------------------------
__global__ void init_last_kernel(int* __restrict__ last_idx) {
  int c = blockIdx.x * blockDim.x + threadIdx.x;
  if (c < NTOK) last_idx[c] = -1;
}

// ---------------- per-row stats: ||h||, ||w_t||, cos, epsilon; last-writer index --
__global__ void __launch_bounds__(128) row_stats_kernel(
    const float* __restrict__ h, const float* __restrict__ enc_w,
    const int* __restrict__ targets, float* __restrict__ n_out,
    float* __restrict__ epsv, int* __restrict__ last_idx) {
  __shared__ float s0[128], s1[128], s2[128];
  const int i = blockIdx.x;
  const int tid = threadIdx.x;
  const int t = targets[i];
  const float* hp = h + (size_t)i * DK;
  const float* wp = enc_w + (size_t)t * DK;
  float hsq = 0.f, dw = 0.f, wsq = 0.f;
  for (int k = tid; k < DK; k += 128) {
    float hv = hp[k], wv = wp[k];
    hsq += hv * hv; dw += hv * wv; wsq += wv * wv;
  }
  s0[tid] = hsq; s1[tid] = dw; s2[tid] = wsq;
  __syncthreads();
  for (int s = 64; s > 0; s >>= 1) {
    if (tid < s) { s0[tid] += s0[tid+s]; s1[tid] += s1[tid+s]; s2[tid] += s2[tid+s]; }
    __syncthreads();
  }
  if (tid == 0) {
    float no = sqrtf(s0[0] + EPSN);
    float nw = sqrtf(s2[0] + EPSN);
    float cosv = s1[0] / (no * nw);
    n_out[i] = no;
    epsv[i]  = (cosv > 0.f) ? ALPHA * nw : 0.f;
    atomicMax(&last_idx[t], i);   // last-write-wins == max index wins
  }
}

// ---------------- noise_outputs[i] = -eps[jl]/||h_jl|| * (h_i . h_jl) ------------
__global__ void __launch_bounds__(128) noise_kernel(
    const float* __restrict__ h, const int* __restrict__ targets,
    const int* __restrict__ last_idx, const float* __restrict__ n_out,
    const float* __restrict__ epsv, float* __restrict__ noise) {
  __shared__ float sd[128];
  const int i = blockIdx.x;
  const int tid = threadIdx.x;
  const int jl = last_idx[targets[i]];   // >= 0 always (targets[i] was written)
  const float* hp = h + (size_t)i  * DK;
  const float* jp = h + (size_t)jl * DK;
  float d = 0.f;
  for (int k = tid; k < DK; k += 128) d += hp[k] * jp[k];
  sd[tid] = d;
  __syncthreads();
  for (int s = 64; s > 0; s >>= 1) {
    if (tid < s) sd[tid] += sd[tid+s];
    __syncthreads();
  }
  if (tid == 0) noise[i] = -epsv[jl] * sd[0] / n_out[jl];
}

// ---------------- pre-convert f32 rows(x DK) -> padded bf16 hi/lo rows(x DKP) ----
__global__ void __launch_bounds__(128) convert_kernel(
    const float* __restrict__ src, __bf16* __restrict__ hi, __bf16* __restrict__ lo) {
  const int row = blockIdx.x;
  const float* sp = src + (size_t)row * DK;
  __bf16* hp = hi + (size_t)row * DKP;
  __bf16* lp = lo + (size_t)row * DKP;
  for (int k = threadIdx.x; k < DKP; k += 128) {
    float v = (k < DK) ? sp[k] : 0.f;
    __bf16 a, b; split2(v, a, b);
    hp[k] = a; lp[k] = b;
  }
}

// ---------------- fragment loaders -----------------------------------------------
__device__ __forceinline__ v16bf load_frag_bf(const __bf16* p) {
  v8bf c0 = *(const v8bf*)(p);
  v8bf c1 = *(const v8bf*)(p + 16);
  v16bf r;
#pragma unroll
  for (int e = 0; e < 8; ++e) { r[e] = c0[e]; r[e + 8] = c1[e]; }
  return r;
}
__device__ __forceinline__ void load_frag_split(const float* p, bool tail,
                                                v16bf& hi, v16bf& lo) {
  v4f a0 = *(const v4f*)(p);
  v4f a1 = *(const v4f*)(p + 4);
  v4f b0 = {0.f,0.f,0.f,0.f}, b1 = {0.f,0.f,0.f,0.f};
  if (!tail) { b0 = *(const v4f*)(p + 16); b1 = *(const v4f*)(p + 20); }
  float vals[16];
#pragma unroll
  for (int e = 0; e < 4; ++e) {
    vals[e] = a0[e]; vals[e+4] = a1[e]; vals[e+8] = b0[e]; vals[e+12] = b1[e];
  }
#pragma unroll
  for (int e = 0; e < 16; ++e) { __bf16 a, b; split2(vals[e], a, b); hi[e] = a; lo[e] = b; }
}

// ---------------- GEMM epilogue for one 16x16 tile -------------------------------
__device__ __forceinline__ void gemm_epilogue(
    v8f acc, int row0, int col0, int mn, int g,
    const float* __restrict__ dec_b, const int* __restrict__ targets,
    const float* __restrict__ noise, float* __restrict__ out) {
  const int col = col0 + mn;
  if (col < NTOK) {
    const float bias = dec_b[col];
#pragma unroll
    for (int r = 0; r < 8; ++r) {
      const int row = row0 + r + 8 * g;        // C layout: VGPR r, lanes16-31 -> M=r+8
      float v = acc[r] + bias;
      if (targets[row] == col) v += noise[row];
      out[(size_t)row * NTOK + col] = v;
    }
  }
}

// ---------------- main GEMM, pre-converted bf16 hi/lo (fast path) ----------------
// grid (393, 140), block 128 = 4 waves. Each wave owns a 16x32 output strip:
// two B-tiles (named vars, no arrays) reuse one A hi/lo fragment pair.
// Per k-step: 12 b128 loads feed 6 WMMAs; acc0/acc1 chains interleaved.
__global__ void __launch_bounds__(128) gemm_fast_kernel(
    const __bf16* __restrict__ ah, const __bf16* __restrict__ al,
    const __bf16* __restrict__ bh, const __bf16* __restrict__ bl,
    const float* __restrict__ dec_b, const int* __restrict__ targets,
    const float* __restrict__ noise, float* __restrict__ out) {
  const int lane = threadIdx.x & 31;
  const int wave = threadIdx.x >> 5;
  const int g  = lane >> 4;
  const int mn = lane & 15;
  const int row0 = blockIdx.y << 4;
  const int col0 = blockIdx.x * 128 + wave * 32;

  int c0 = col0 + mn;      if (c0 > NTOK - 1) c0 = NTOK - 1;  // clamp loads, mask stores
  int c1 = col0 + 16 + mn; if (c1 > NTOK - 1) c1 = NTOK - 1;

  const __bf16* ahp  = ah + (size_t)(row0 + mn) * DKP + 8 * g;
  const __bf16* alp  = al + (size_t)(row0 + mn) * DKP + 8 * g;
  const __bf16* bhp0 = bh + (size_t)c0 * DKP + 8 * g;
  const __bf16* blp0 = bl + (size_t)c0 * DKP + 8 * g;
  const __bf16* bhp1 = bh + (size_t)c1 * DKP + 8 * g;
  const __bf16* blp1 = bl + (size_t)c1 * DKP + 8 * g;

  v8f acc0 = {0.f,0.f,0.f,0.f,0.f,0.f,0.f,0.f};
  v8f acc1 = {0.f,0.f,0.f,0.f,0.f,0.f,0.f,0.f};
#pragma unroll
  for (int ks = 0; ks < KSTEPS; ++ks) {
    const int k0 = ks * 32;
    const v16bf fah  = load_frag_bf(ahp  + k0);
    const v16bf fal  = load_frag_bf(alp  + k0);
    const v16bf fbh0 = load_frag_bf(bhp0 + k0);
    const v16bf fbl0 = load_frag_bf(blp0 + k0);
    const v16bf fbh1 = load_frag_bf(bhp1 + k0);
    const v16bf fbl1 = load_frag_bf(blp1 + k0);
    acc0 = __builtin_amdgcn_wmma_f32_16x16x32_bf16(false, fah, false, fbh0, (short)0, acc0, false, false);
    acc1 = __builtin_amdgcn_wmma_f32_16x16x32_bf16(false, fah, false, fbh1, (short)0, acc1, false, false);
    acc0 = __builtin_amdgcn_wmma_f32_16x16x32_bf16(false, fah, false, fbl0, (short)0, acc0, false, false);
    acc1 = __builtin_amdgcn_wmma_f32_16x16x32_bf16(false, fah, false, fbl1, (short)0, acc1, false, false);
    acc0 = __builtin_amdgcn_wmma_f32_16x16x32_bf16(false, fal, false, fbh0, (short)0, acc0, false, false);
    acc1 = __builtin_amdgcn_wmma_f32_16x16x32_bf16(false, fal, false, fbh1, (short)0, acc1, false, false);
  }
  gemm_epilogue(acc0, row0, col0,      mn, g, dec_b, targets, noise, out);
  gemm_epilogue(acc1, row0, col0 + 16, mn, g, dec_b, targets, noise, out);
}

// ---------------- main GEMM, on-the-fly split (fallback if ws too small) ---------
__global__ void __launch_bounds__(128) gemm_slow_kernel(
    const float* __restrict__ h, const float* __restrict__ dec_w,
    const float* __restrict__ dec_b, const int* __restrict__ targets,
    const float* __restrict__ noise, float* __restrict__ out) {
  const int lane = threadIdx.x & 31;
  const int wave = threadIdx.x >> 5;
  const int g  = lane >> 4;
  const int mn = lane & 15;
  const int row0 = blockIdx.y << 4;
  const int col0 = blockIdx.x * 64 + wave * 16;
  int cl = col0 + mn; if (cl > NTOK - 1) cl = NTOK - 1;

  const float* ap = h + (size_t)(row0 + mn) * DK + 8 * g;
  const float* bp = dec_w + (size_t)cl * DK + 8 * g;

  v8f acc = {0.f,0.f,0.f,0.f,0.f,0.f,0.f,0.f};
  for (int ks = 0; ks < KSTEPS; ++ks) {
    const bool tail = (ks == KSTEPS - 1);
    const int k0 = ks * 32;
    v16bf fah, fal, fbh, fbl;
    load_frag_split(ap + k0, tail, fah, fal);
    load_frag_split(bp + k0, tail, fbh, fbl);
    acc = __builtin_amdgcn_wmma_f32_16x16x32_bf16(false, fah, false, fbh, (short)0, acc, false, false);
    acc = __builtin_amdgcn_wmma_f32_16x16x32_bf16(false, fah, false, fbl, (short)0, acc, false, false);
    acc = __builtin_amdgcn_wmma_f32_16x16x32_bf16(false, fal, false, fbh, (short)0, acc, false, false);
  }
  gemm_epilogue(acc, row0, col0, mn, g, dec_b, targets, noise, out);
}

// ---------------- per-row online max + log-sum-exp -------------------------------
__global__ void __launch_bounds__(256) lse_kernel(const float* __restrict__ out,
                                                  float* __restrict__ shift) {
  __shared__ float smx[256], ssm[256];
  const int row = blockIdx.x;
  const int tid = threadIdx.x;
  const float* p = out + (size_t)row * NTOK;
  float mx = -3.4e38f, sum = 0.f;
  for (int c = tid; c < NTOK; c += 256) {
    float v = p[c];
    if (v > mx) { sum = sum * expf(mx - v) + 1.f; mx = v; }
    else sum += expf(v - mx);
  }
  smx[tid] = mx; ssm[tid] = sum;
  __syncthreads();
  for (int s = 128; s > 0; s >>= 1) {
    if (tid < s) {
      float m1 = smx[tid], m2 = smx[tid+s];
      float M = fmaxf(m1, m2);
      ssm[tid] = ssm[tid] * expf(m1 - M) + ssm[tid+s] * expf(m2 - M);
      smx[tid] = M;
    }
    __syncthreads();
  }
  if (tid == 0) shift[row] = smx[0] + logf(ssm[0]);
}

__global__ void __launch_bounds__(256) norm_kernel(float* __restrict__ out,
                                                   const float* __restrict__ shift) {
  const int row = blockIdx.y;
  const int col = blockIdx.x * 256 + threadIdx.x;
  if (col < NTOK) out[(size_t)row * NTOK + col] -= shift[row];
}

// ---------------- launcher -------------------------------------------------------
extern "C" void kernel_launch(void* const* d_in, const int* in_sizes, int n_in,
                              void* d_out, int out_size, void* d_ws, size_t ws_size,
                              hipStream_t stream) {
  const float* x      = (const float*)d_in[0];   // (35,64,400) -> (2240,400)
  const float* dec_w  = (const float*)d_in[1];   // (50257,400)
  const float* dec_b  = (const float*)d_in[2];   // (50257,)
  const float* enc_w  = (const float*)d_in[3];   // (50257,400)
  const int*   targets= (const int*)  d_in[4];   // (2240,)
  float* out = (float*)d_out;
  (void)in_sizes; (void)n_in; (void)out_size;

  // ---- workspace layout ----
  char* ws = (char*)d_ws;
  int*   last_idx = (int*)ws;                 size_t off = 50272 * sizeof(int); // 201088, 16B-aligned
  float* n_out = (float*)(ws + off);          off += NROW * sizeof(float);
  float* epsv  = (float*)(ws + off);          off += NROW * sizeof(float);
  float* noise = (float*)(ws + off);          off += NROW * sizeof(float);
  float* shift = (float*)(ws + off);          off += NROW * sizeof(float);
  const size_t szA = (size_t)NROW * DKP * sizeof(__bf16);   // 1.86 MB each
  const size_t szB = (size_t)NTOK * DKP * sizeof(__bf16);   // 41.8 MB each
  __bf16* ahb = (__bf16*)(ws + off);
  __bf16* alb = (__bf16*)(ws + off + szA);
  __bf16* bhb = (__bf16*)(ws + off + 2 * szA);
  __bf16* blb = (__bf16*)(ws + off + 2 * szA + szB);
  const size_t need_fast = off + 2 * szA + 2 * szB;         // ~87.6 MB
  const bool fast = (ws_size >= need_fast);                  // deterministic per run

  // ---- pipeline ----
  init_last_kernel<<<(NTOK + 255) / 256, 256, 0, stream>>>(last_idx);
  row_stats_kernel<<<NROW, 128, 0, stream>>>(x, enc_w, targets, n_out, epsv, last_idx);
  noise_kernel<<<NROW, 128, 0, stream>>>(x, targets, last_idx, n_out, epsv, noise);

  if (fast) {
    convert_kernel<<<NROW, 128, 0, stream>>>(x, ahb, alb);
    convert_kernel<<<NTOK, 128, 0, stream>>>(dec_w, bhb, blb);
    const dim3 ggrid((NTOK + 127) / 128, NROW / 16);   // (393, 140)
    gemm_fast_kernel<<<ggrid, 128, 0, stream>>>(ahb, alb, bhb, blb, dec_b, targets, noise, out);
  } else {
    const dim3 ggrid((NTOK + 63) / 64, NROW / 16);     // (786, 140)
    gemm_slow_kernel<<<ggrid, 128, 0, stream>>>(x, dec_w, dec_b, targets, noise, out);
  }

  lse_kernel<<<NROW, 256, 0, stream>>>(out, shift);
  norm_kernel<<<dim3((NTOK + 255) / 256, NROW), 256, 0, stream>>>(out, shift);
}
